// Model_21620865368621
// MI455X (gfx1250) — compile-verified, matrix operand order checked by reading, not run on hardware
//
#include <hip/hip_runtime.h>
#include <hip/hip_bf16.h>

// RNNT + CTC loss, fused for MI455X (gfx1250, wave32, WMMA).
//
// Dominant work: joiner GEMM [66560 x 256] x [256 x 1024] -> row-wise
// logsumexp only. We fuse the GEMM with an online logsumexp so the
// [B,T,U+1,V] tensor (273 MB fp32) is never materialized. GEMM runs in
// f16 WMMA (v_wmma_f32_16x16x32_f16) with fp32 accumulation; inputs are
// tanh outputs in [-1,1] so f16 is numerically safe for K=256 dots.

#define B_  4
#define T_  256
#define U_  64
#define V_  1024
#define D_  256
#define H_  256
#define S_  129           // 2U+1
#define BLANK_ 1023
#define NEGV (-1.0e30f)

typedef _Float16 v16h __attribute__((ext_vector_type(16)));
typedef _Float16 v8h  __attribute__((ext_vector_type(8)));
typedef float    v8f  __attribute__((ext_vector_type(8)));

__device__ __forceinline__ float la2(float a, float b) {
    float m = fmaxf(a, b), n = fminf(a, b);
    return m + log1pf(__expf(n - m));
}
__device__ __forceinline__ float la3(float a, float b, float c) {
    return la2(la2(a, b), c);
}

// ---------------------------------------------------------------------------
// Kernel 0: pack W_out (fp32 [H,V]) into f16 B-fragment layout.
// Layout: w_pack[((nt*8 + kt)*32 + lane)*16 + h]  (32 bytes contiguous/lane)
//   lane 0-15 :  h 0-7 -> K = kt*32 + h      ; h 8-15 -> K = kt*32 + h + 8
//   lane 16-31:  h 0-7 -> K = kt*32 + h + 8  ; h 8-15 -> K = kt*32 + h + 16
//   N = nt*16 + (lane & 15)
// ---------------------------------------------------------------------------
__global__ __launch_bounds__(256) void pack_wout_kernel(
    const float* __restrict__ W_out, _Float16* __restrict__ w_pack) {
    int idx  = blockIdx.x * 256 + threadIdx.x;     // 0 .. 262143
    int hh   = idx & 15;
    int lane = (idx >> 4) & 31;
    int kt   = (idx >> 9) & 7;
    int nt   = idx >> 12;
    int hiL  = lane >> 4;
    int kk   = (hh < 8 ? hh : hh + 8) + (hiL ? 8 : 0);
    int K    = kt * 32 + kk;
    int N    = nt * 16 + (lane & 15);
    w_pack[idx] = (_Float16)W_out[K * V_ + N];
}

// ---------------------------------------------------------------------------
// Kernel 1: row-projection  out[row, :256] = X[row, :K] @ W[K,256] + bias
// One block per row, 256 threads (one output column each).
// ---------------------------------------------------------------------------
__global__ __launch_bounds__(256) void proj_kernel(
    const float* __restrict__ X, const float* __restrict__ W,
    const float* __restrict__ bias, float* __restrict__ out, int K) {
    int row = blockIdx.x, tid = threadIdx.x;
    __shared__ float xrow[256];
    xrow[tid] = (tid < K) ? X[row * K + tid] : 0.f;
    __syncthreads();
    float s = bias[tid];
    for (int k = 0; k < K; ++k) s += xrow[k] * W[k * 256 + tid];
    out[row * 256 + tid] = s;
}

// ---------------------------------------------------------------------------
// Kernel 2: CTC head.  For each (b,t): logits = x_enc @ W_ctc + b_ctc,
// lse over V, emit[s] = logit[ext[s]] - lse for the S=129 extended labels.
// ---------------------------------------------------------------------------
__global__ __launch_bounds__(256) void ctc_head_kernel(
    const float* __restrict__ x_enc, const float* __restrict__ W_ctc,
    const float* __restrict__ b_ctc, const int* __restrict__ target,
    float* __restrict__ emit) {
    int bt = blockIdx.x, b = bt >> 8, tid = threadIdx.x;
    __shared__ float xrow[256];
    __shared__ float lg[1024];
    __shared__ float red[256];

    xrow[tid] = x_enc[bt * 256 + tid];
    __syncthreads();

    for (int c = tid; c < 1024; c += 256) {
        float s = b_ctc[c];
        for (int k = 0; k < 256; ++k) s += xrow[k] * W_ctc[k * 1024 + c];
        lg[c] = s;
    }
    __syncthreads();

    float m = -3.0e38f;
    for (int c = tid; c < 1024; c += 256) m = fmaxf(m, lg[c]);
    red[tid] = m;
    __syncthreads();
    for (int st = 128; st > 0; st >>= 1) {
        if (tid < st) red[tid] = fmaxf(red[tid], red[tid + st]);
        __syncthreads();
    }
    m = red[0];
    __syncthreads();
    float s = 0.f;
    for (int c = tid; c < 1024; c += 256) s += __expf(lg[c] - m);
    red[tid] = s;
    __syncthreads();
    for (int st = 128; st > 0; st >>= 1) {
        if (tid < st) red[tid] += red[tid + st];
        __syncthreads();
    }
    float lse = m + __logf(red[0]);

    if (tid < S_) {
        int lab = (tid & 1) ? target[b * U_ + (tid >> 1)] : BLANK_;
        emit[bt * S_ + tid] = lg[lab] - lse;
    }
}

// ---------------------------------------------------------------------------
// Kernel 3: joiner (the big one).  One block per (b,t); M = u-rows (65,
// padded to 80 = 5 WMMA tiles), N = V = 1024 (64 tiles), K = H = 256 (8
// tiles of 32).  8 waves; wave w owns n-tiles {i*8+w}.  Online logsumexp
// per row fused into the GEMM epilogue; blank/label log-probs recomputed
// with cheap fp32 dots afterwards.
// ---------------------------------------------------------------------------
__global__ __launch_bounds__(256) void joiner_kernel(
    const float* __restrict__ enc_proj,      // [B*T, H]
    const float* __restrict__ dec_proj,      // [B*(U+1), H]
    const _Float16* __restrict__ w_pack,     // packed f16 W_out fragments
    const float* __restrict__ b_out,         // [V]
    const float* __restrict__ W_out,         // [H, V] fp32 (for dots)
    const int* __restrict__ target,          // [B, U]
    float* __restrict__ lp_blank,            // [B*T, U+1]
    float* __restrict__ lp_label) {          // [B*T, U]
    const int bt   = blockIdx.x;
    const int b    = bt >> 8;
    const int tid  = threadIdx.x;
    const int lane = tid & 31;
    const int w    = tid >> 5;
    const int ln16 = lane & 15;
    const int hi   = lane >> 4;

    __shared__ _Float16 hsh[80][264];     // padded stride: conflict-free b128
    __shared__ float bout_sh[1024];
    __shared__ float part_mx[8][80];
    __shared__ float part_sm[8][80];
    __shared__ float lse_sh[80];

    // Build h(u,k) = tanh(e_k + d_{u,k}) as f16 in LDS (thread = column k).
    {
        float e = enc_proj[bt * 256 + tid];
        for (int u = 0; u < 65; ++u) {
            float dv = dec_proj[(b * 65 + u) * 256 + tid];
            hsh[u][tid] = (_Float16)tanhf(e + dv);
        }
        for (int u = 65; u < 80; ++u) hsh[u][tid] = (_Float16)0.f;
        for (int i = tid; i < 1024; i += 256) bout_sh[i] = b_out[i];
    }
    __syncthreads();

    // Per-lane online (max, sumexp) stats for all 5 M-tiles.
    float mx[5][8], sm[5][8];
#pragma unroll
    for (int m = 0; m < 5; ++m)
#pragma unroll
        for (int r = 0; r < 8; ++r) { mx[m][r] = -3.0e38f; sm[m][r] = 0.f; }

#pragma unroll 1
    for (int i = 0; i < 8; ++i) {
        const int nt = i * 8 + w;
        // B fragments for this n-tile, all 8 k-tiles (64 VGPRs).
        v16h bf[8];
#pragma unroll
        for (int k = 0; k < 8; ++k)
            bf[k] = *(const v16h*)(w_pack + (((nt * 8 + k) * 32 + lane) << 4));
        const float bo = bout_sh[nt * 16 + ln16];

#pragma unroll
        for (int m = 0; m < 5; ++m) {
            v8f acc = {0.f, 0.f, 0.f, 0.f, 0.f, 0.f, 0.f, 0.f};
#pragma unroll
            for (int k = 0; k < 8; ++k) {
                // A fragment: lane holds row m*16 + ln16, two 16B K-chunks
                // loaded straight into the halves of one v16h (no shuffles).
                const _Float16* arow = &hsh[m * 16 + ln16][k * 32];
                union { v16h v; v8h h2[2]; } au;
                au.h2[0] = *(const v8h*)(arow + hi * 8);
                au.h2[1] = *(const v8h*)(arow + 16 + hi * 8);
                acc = __builtin_amdgcn_wmma_f32_16x16x32_f16(
                    false, au.v, false, bf[k], (short)0, acc, false, false);
            }
#pragma unroll
            for (int r = 0; r < 8; ++r) {
                float x  = acc[r] + bo;
                float nm = fmaxf(mx[m][r], x);
                sm[m][r] = sm[m][r] * __expf(mx[m][r] - nm) + __expf(x - nm);
                mx[m][r] = nm;
            }
        }
    }

    // Reduce stats across the 16 lanes sharing each row (columns differ).
#pragma unroll
    for (int m = 0; m < 5; ++m)
#pragma unroll
        for (int r = 0; r < 8; ++r) {
            float tm = mx[m][r], ts = sm[m][r];
#pragma unroll
            for (int off = 8; off >= 1; off >>= 1) {
                float om = __shfl_xor(tm, off, 16);
                float os = __shfl_xor(ts, off, 16);
                float nm = fmaxf(tm, om);
                ts = ts * __expf(tm - nm) + os * __expf(om - nm);
                tm = nm;
            }
            if (ln16 == 0) {
                int row = m * 16 + hi * 8 + r;   // C-layout: VGPR r -> M = r (+8)
                part_mx[w][row] = tm;
                part_sm[w][row] = ts;
            }
        }
    __syncthreads();

    // Combine the 8 per-wave partials -> lse per row.
    if (tid < 80) {
        float tm = part_mx[0][tid], ts = part_sm[0][tid];
#pragma unroll
        for (int ww = 1; ww < 8; ++ww) {
            float om = part_mx[ww][tid], os = part_sm[ww][tid];
            float nm = fmaxf(tm, om);
            ts = ts * __expf(tm - nm) + os * __expf(om - nm);
            tm = nm;
        }
        lse_sh[tid] = tm + __logf(ts);
    }
    __syncthreads();

    // Blank (u=0..64) and label (u=0..63) log-probs via cheap fp32 dots.
    if (tid < 129) {
        int isblank = tid < 65;
        int u = isblank ? tid : tid - 65;
        int v = isblank ? BLANK_ : target[b * U_ + u];
        float s = 0.f;
        for (int k = 0; k < 256; ++k)
            s += (float)hsh[u][k] * W_out[k * V_ + v];
        float val = s + bout_sh[v] - lse_sh[u];
        if (isblank) lp_blank[bt * 65 + u] = val;
        else         lp_label[bt * 64 + u] = val;
    }
}

// ---------------------------------------------------------------------------
// Kernel 4: RNNT forward DP, anti-diagonal wavefront over the (T, U+1)
// lattice.  One block per batch, two LDS diagonals.
// ---------------------------------------------------------------------------
__global__ __launch_bounds__(96) void rnnt_dp_kernel(
    const float* __restrict__ lp_blank, const float* __restrict__ lp_label,
    const int* __restrict__ frame_length, const int* __restrict__ target_length,
    float* __restrict__ nll_rnnt) {
    int b = blockIdx.x, tid = threadIdx.x;
    __shared__ float diag[2][65];
    __shared__ float llsh;
    int ft = frame_length[b], tl = target_length[b];
    const float* blk = lp_blank + b * T_ * 65;
    const float* lbl = lp_label + b * T_ * 64;

    for (int d = 0; d < T_ + U_; ++d) {
        int u = tid;
        if (u <= U_) {
            int t = d - u;
            if (t >= 0 && t < T_) {
                float a;
                if (t == 0 && u == 0) a = 0.f;
                else if (t == 0)
                    a = diag[(d - 1) & 1][u - 1] + lbl[(u - 1)];
                else if (u == 0)
                    a = diag[(d - 1) & 1][0] + blk[(t - 1) * 65];
                else {
                    float p1 = diag[(d - 1) & 1][u]     + blk[(t - 1) * 65 + u];
                    float p2 = diag[(d - 1) & 1][u - 1] + lbl[t * 64 + (u - 1)];
                    a = la2(p1, p2);
                }
                diag[d & 1][u] = a;
                if (t == ft - 1 && u == tl) llsh = a + blk[t * 65 + u];
            }
        }
        __syncthreads();
    }
    if (tid == 0) nll_rnnt[b] = -llsh;
}

// ---------------------------------------------------------------------------
// Kernel 5: CTC forward DP.  One block per batch, S=129 states in parallel,
// serial over T with LDS ping-pong.
// ---------------------------------------------------------------------------
__global__ __launch_bounds__(160) void ctc_dp_kernel(
    const float* __restrict__ emit, const int* __restrict__ target,
    const int* __restrict__ frame_length, const int* __restrict__ target_length,
    float* __restrict__ nll_ctc) {
    int b = blockIdx.x, tid = threadIdx.x;
    __shared__ float buf[2][S_];
    __shared__ unsigned char skipsh[S_];
    int ft = frame_length[b], tl = target_length[b];
    const float* em = emit + b * T_ * S_;

    if (tid < S_) {
        float init = NEGV;
        if (tid == 0) init = em[0];
        if (tid == 1) init = em[1];
        buf[0][tid] = init;
        bool sk = false;
        if (tid >= 3 && (tid & 1))
            sk = (target[b * U_ + (tid >> 1)] != target[b * U_ + (tid >> 1) - 1]);
        skipsh[tid] = sk ? 1 : 0;
    }
    __syncthreads();

    for (int t = 1; t < T_; ++t) {
        int pp = (t - 1) & 1, cc = t & 1;
        if (tid < S_) {
            float prev = buf[pp][tid];
            float nv;
            if (t < ft) {
                float p1 = prev;
                float p2 = (tid >= 1) ? buf[pp][tid - 1] : NEGV;
                float p3 = (tid >= 2 && skipsh[tid]) ? buf[pp][tid - 2] : NEGV;
                nv = em[t * S_ + tid] + la3(p1, p2, p3);
            } else {
                nv = prev;
            }
            buf[cc][tid] = nv;
        }
        __syncthreads();
    }
    if (tid == 0) {
        int sl = 2 * tl;
        float cl = la2(buf[(T_ - 1) & 1][sl], buf[(T_ - 1) & 1][sl - 1]);
        nll_ctc[b] = (cl < NEGV * 0.5f) ? 0.f : -cl;
    }
}

// ---------------------------------------------------------------------------
// Kernel 6: finalize  loss = mean(nll_rnnt) + 0.3 * sum(nll_ctc)/B
// ---------------------------------------------------------------------------
__global__ void finalize_kernel(const float* __restrict__ nll_rnnt,
                                const float* __restrict__ nll_ctc,
                                float* __restrict__ out) {
    if (threadIdx.x == 0 && blockIdx.x == 0) {
        float r = 0.f, c = 0.f;
        for (int b = 0; b < B_; ++b) { r += nll_rnnt[b]; c += nll_ctc[b]; }
        out[0] = r * 0.25f + 0.3f * (c * 0.25f);
    }
}

// ---------------------------------------------------------------------------
extern "C" void kernel_launch(void* const* d_in, const int* in_sizes, int n_in,
                              void* d_out, int out_size, void* d_ws, size_t ws_size,
                              hipStream_t stream) {
    (void)in_sizes; (void)n_in; (void)out_size; (void)ws_size;
    const float* x_enc = (const float*)d_in[0];
    const float* x_dec = (const float*)d_in[1];
    const float* W_enc = (const float*)d_in[2];
    const float* b_enc = (const float*)d_in[3];
    const float* W_dec = (const float*)d_in[4];
    const float* b_dec = (const float*)d_in[5];
    const float* W_out = (const float*)d_in[6];
    const float* b_out = (const float*)d_in[7];
    const float* W_ctc = (const float*)d_in[8];
    const float* b_ctc = (const float*)d_in[9];
    const int* target        = (const int*)d_in[10];
    const int* frame_length  = (const int*)d_in[11];
    const int* target_length = (const int*)d_in[12];

    char* ws = (char*)d_ws;
    size_t off = 0;
    auto alloc = [&](size_t bytes) -> void* {
        void* p = ws + off;
        off = (off + bytes + 255) & ~(size_t)255;
        return p;
    };
    float*     enc_proj = (float*)alloc((size_t)B_ * T_ * H_ * 4);        // 1 MB
    float*     dec_proj = (float*)alloc((size_t)B_ * (U_ + 1) * H_ * 4);  // 266 KB
    _Float16*  w_pack   = (_Float16*)alloc((size_t)H_ * V_ * 2);          // 512 KB
    float*     emit     = (float*)alloc((size_t)B_ * T_ * S_ * 4);        // 528 KB
    float*     lp_blank = (float*)alloc((size_t)B_ * T_ * (U_ + 1) * 4);  // 266 KB
    float*     lp_label = (float*)alloc((size_t)B_ * T_ * U_ * 4);        // 262 KB
    float*     nll_rnnt = (float*)alloc(64);
    float*     nll_ctc  = (float*)alloc(64);

    pack_wout_kernel<<<1024, 256, 0, stream>>>(W_out, w_pack);
    proj_kernel<<<B_ * T_, 256, 0, stream>>>(x_enc, W_enc, b_enc, enc_proj, D_);
    proj_kernel<<<B_ * (U_ + 1), 256, 0, stream>>>(x_dec, W_dec, b_dec, dec_proj, H_);
    ctc_head_kernel<<<B_ * T_, 256, 0, stream>>>(x_enc, W_ctc, b_ctc, target, emit);
    joiner_kernel<<<B_ * T_, 256, 0, stream>>>(enc_proj, dec_proj, w_pack, b_out,
                                               W_out, target, lp_blank, lp_label);
    rnnt_dp_kernel<<<B_, 96, 0, stream>>>(lp_blank, lp_label, frame_length,
                                          target_length, nll_rnnt);
    ctc_dp_kernel<<<B_, 160, 0, stream>>>(emit, target, frame_length,
                                          target_length, nll_ctc);
    finalize_kernel<<<1, 32, 0, stream>>>(nll_rnnt, nll_ctc, (float*)d_out);
}
